// FindArgModel_79697413144906
// MI455X (gfx1250) — matrix operand order repaired and checked by reading, not compiled, not run on hardware
//
#include <hip/hip_runtime.h>
#include <math.h>

// ---------------- problem constants ----------------
#define Bc   512
#define Sc   128
#define Hc   512
#define H3c  (3*Hc)
#define NROWS ((Sc+1)*Bc)      // 66048 rows fed to the likelihood MLP
#define EOS_TOK 1

// ---------------- CDNA5 WMMA types ----------------
typedef __attribute__((ext_vector_type(16))) __bf16 v16bf;
typedef __attribute__((ext_vector_type(8)))  float  v8f;

__device__ __forceinline__ v8f wmma_bf16(v16bf a, v16bf b, v8f c) {
    // v_wmma_f32_16x16x32_bf16  D = A*B + C
    return __builtin_amdgcn_wmma_f32_16x16x32_bf16(
        false, a, false, b, (short)0, c, false, false);
}

__device__ __forceinline__ __bf16 f2bf(float f) {
    unsigned u = __builtin_bit_cast(unsigned, f);
    unsigned r = u + 0x7FFFu + ((u >> 16) & 1u);     // round-to-nearest-even
    unsigned short h = (unsigned short)(r >> 16);
    return __builtin_bit_cast(__bf16, h);
}

__device__ __forceinline__ float sigmoidf_(float x) {
    return 1.0f / (1.0f + expf(-x));
}

// A operand: 16x32 bf16 tile. lane holds row M=lane%16; 16 contiguous K
// {kb + half*16 .. +15} (matches B operand order)
__device__ __forceinline__ v16bf load_a_bf16(const __bf16* __restrict__ row,
                                             int kb, int half) {
    union { v16bf v; uint4 q[2]; } u;
    const uint4* p = reinterpret_cast<const uint4*>(row + kb + half * 16);
    u.q[0] = p[0];      // 8 bf16
    u.q[1] = p[1];      // next 8 bf16
    return u.v;
}

// B operand: 32x16 bf16. lane holds col N=lane%16 (a W row, contiguous in K);
// 16 contiguous K starting at kb + (lane/16)*16.
__device__ __forceinline__ v16bf load_b_bf16(const __bf16* __restrict__ wrow,
                                             int kb, int hk) {
    union { v16bf v; uint4 q[2]; } u;
    const uint4* p = reinterpret_cast<const uint4*>(wrow + kb + hk);
    u.q[0] = p[0];
    u.q[1] = p[1];
    return u.v;
}

// ---------------- prep kernels ----------------
__global__ void k_cvt_bf16(const float* __restrict__ src,
                           __bf16* __restrict__ dst, int n) {
    int i = blockIdx.x * blockDim.x + threadIdx.x;
    if (i < n) dst[i] = f2bf(src[i]);
}

__global__ void k_embed_x(const int* __restrict__ goal,
                          const float* __restrict__ wemb,
                          __bf16* __restrict__ xbuf) {
    const long long total = (long long)Sc * Bc * Hc;
    long long stride = (long long)gridDim.x * blockDim.x;
    for (long long e = (long long)blockIdx.x * blockDim.x + threadIdx.x;
         e < total; e += stride) {
        int k = (int)(e % Hc);
        long long row = e / Hc;
        int b = (int)(row % Bc);
        int s = (int)(row / Bc);
        int tok = goal[b * Sc + s];
        float v = wemb[(long long)tok * Hc + k];
        xbuf[row * Hc + k] = f2bf(v > 0.f ? v : 0.f);
    }
}

__global__ void k_init_h0_eos(const int* __restrict__ stem,
                              const float* __restrict__ semb,
                              const float* __restrict__ wemb,
                              float* __restrict__ h0,
                              __bf16* __restrict__ h0bf,
                              __bf16* __restrict__ xeos) {
    int i = blockIdx.x * blockDim.x + threadIdx.x;
    if (i < Bc * Hc) {
        int b = i / Hc, k = i % Hc;
        float v = semb[(long long)stem[b] * Hc + k];
        h0[i] = v;
        h0bf[i] = f2bf(v);
    }
    if (i < Hc) {
        float v = wemb[(long long)EOS_TOK * Hc + i];
        xeos[i] = f2bf(v > 0.f ? v : 0.f);
    }
}

// ---------------- fused GRU step (WMMA) ----------------
// grid = (B/16, H/16/8), block = 256 (8 waves); wave owns a 16x16 (b,j) tile.
// K loop covers h-part (W_hh) and x-part (W_ih) in the same pass:
//   accR/accZ combined, gate-n split into x-part and h-part (r * hn rule).
// h is kept in fp32 (recurrence precision) AND bf16 (zero-VALU A-operand loads
// on the serial critical path).
__global__ __launch_bounds__(256)
void k_gru_step(const float* __restrict__ h_in, const __bf16* __restrict__ h_in_bf,
                float* __restrict__ h_out, __bf16* __restrict__ h_out_bf,
                const __bf16* __restrict__ xbase, int xstride,
                const __bf16* __restrict__ Wih, const __bf16* __restrict__ Whh,
                const float* __restrict__ b_ih, const float* __restrict__ b_hh,
                __bf16* __restrict__ hcat, int orow_base) {
    const int lane = threadIdx.x & 31;
    const int wave = threadIdx.x >> 5;
    const int l16  = lane & 15;
    const int half = lane >> 4;
    const int hk   = half * 16;
    const int b0   = blockIdx.x * 16;
    const int j0   = (blockIdx.y * 8 + wave) * 16;
    const int j    = j0 + l16;

    const __bf16* hrow = h_in_bf + (size_t)(b0 + l16) * Hc;
    const __bf16* xrow = xbase + (size_t)l16 * xstride;   // xstride==0 -> EOS broadcast

    const __bf16* wih_r = Wih + (size_t)(0 * Hc + j) * Hc;
    const __bf16* wih_z = Wih + (size_t)(1 * Hc + j) * Hc;
    const __bf16* wih_n = Wih + (size_t)(2 * Hc + j) * Hc;
    const __bf16* whh_r = Whh + (size_t)(0 * Hc + j) * Hc;
    const __bf16* whh_z = Whh + (size_t)(1 * Hc + j) * Hc;
    const __bf16* whh_n = Whh + (size_t)(2 * Hc + j) * Hc;

    v8f accR = {}, accZ = {}, accNx = {}, accNh = {};

#pragma unroll 4
    for (int kc = 0; kc < Hc; kc += 32) {
        __builtin_prefetch(whh_r + kc + 128, 0, 3);   // global_prefetch_b8, near
        v16bf ah = load_a_bf16(hrow, kc, half);
        v16bf ax = load_a_bf16(xrow, kc, half);
        accR  = wmma_bf16(ah, load_b_bf16(whh_r, kc, hk), accR);
        accR  = wmma_bf16(ax, load_b_bf16(wih_r, kc, hk), accR);
        accZ  = wmma_bf16(ah, load_b_bf16(whh_z, kc, hk), accZ);
        accZ  = wmma_bf16(ax, load_b_bf16(wih_z, kc, hk), accZ);
        accNh = wmma_bf16(ah, load_b_bf16(whh_n, kc, hk), accNh);
        accNx = wmma_bf16(ax, load_b_bf16(wih_n, kc, hk), accNx);
    }

    const float bihR = b_ih[j],          bhhR = b_hh[j];
    const float bihZ = b_ih[Hc + j],     bhhZ = b_hh[Hc + j];
    const float bihN = b_ih[2 * Hc + j], bhhN = b_hh[2 * Hc + j];

#pragma unroll
    for (int i = 0; i < 8; ++i) {
        int m = i + half * 8;                 // D layout: M = i + 8*(lane>=16)
        int b = b0 + m;
        float hprev = h_in[(size_t)b * Hc + j];
        float r  = sigmoidf_(accR[i] + bihR + bhhR);
        float z  = sigmoidf_(accZ[i] + bihZ + bhhZ);
        float nt = tanhf((accNx[i] + bihN) + r * (accNh[i] + bhhN));
        float hn = (1.f - z) * nt + z * hprev;
        h_out[(size_t)b * Hc + j]    = hn;
        h_out_bf[(size_t)b * Hc + j] = f2bf(hn);
        hcat[(size_t)(orow_base + b) * Hc + j] = f2bf(hn > 0.f ? hn : 0.f);
    }
}

// ---------------- MLP layer 1 (WMMA GEMM + relu) ----------------
__global__ __launch_bounds__(256)
void k_mlp_in(const __bf16* __restrict__ hcat, const __bf16* __restrict__ Winb,
              const float* __restrict__ in_b, __bf16* __restrict__ a1) {
    const int lane = threadIdx.x & 31;
    const int wave = threadIdx.x >> 5;
    const int l16  = lane & 15;
    const int half = lane >> 4;
    const int hk   = half * 16;
    const int r0   = blockIdx.x * 16;
    const int j    = (blockIdx.y * 8 + wave) * 16 + l16;

    const __bf16* arow = hcat + (size_t)(r0 + l16) * Hc;
    const __bf16* brow = Winb + (size_t)j * Hc;

    v8f acc = {};
#pragma unroll
    for (int kc = 0; kc < Hc; kc += 32)
        acc = wmma_bf16(load_a_bf16(arow, kc, half),
                        load_b_bf16(brow, kc, hk), acc);

    const float bias = in_b[j];
#pragma unroll
    for (int i = 0; i < 8; ++i) {
        int m = i + half * 8;
        float v = acc[i] + bias;
        a1[(size_t)(r0 + m) * Hc + j] = f2bf(v > 0.f ? v : 0.f);
    }
}

// ---------------- MLP layer 2 + output dot, fused ----------------
// block owns 16 rows; 8 waves x 4 j-tiles cover all 512 cols of a2;
// relu(a2)*out_W reduced per-row via LDS ds_add_f32.
__global__ __launch_bounds__(256)
void k_mlp_out(const __bf16* __restrict__ a1, const __bf16* __restrict__ Wl0b,
               const float* __restrict__ l0_b, const float* __restrict__ out_W,
               const float* __restrict__ out_b, float* __restrict__ logits) {
    __shared__ float red[16];
    const int lane = threadIdx.x & 31;
    const int wave = threadIdx.x >> 5;
    const int l16  = lane & 15;
    const int half = lane >> 4;
    const int hk   = half * 16;
    const int r0   = blockIdx.x * 16;

    if (threadIdx.x < 16) red[threadIdx.x] = 0.f;
    __syncthreads();

    const __bf16* arow = a1 + (size_t)(r0 + l16) * Hc;

    float partial[8];
#pragma unroll
    for (int i = 0; i < 8; ++i) partial[i] = 0.f;

    for (int t = 0; t < 4; ++t) {
        const int j = (wave * 4 + t) * 16 + l16;
        const __bf16* brow = Wl0b + (size_t)j * Hc;
        v8f acc = {};
#pragma unroll
        for (int kc = 0; kc < Hc; kc += 32)
            acc = wmma_bf16(load_a_bf16(arow, kc, half),
                            load_b_bf16(brow, kc, hk), acc);
        const float bias = l0_b[j];
        const float ow   = out_W[j];
#pragma unroll
        for (int i = 0; i < 8; ++i) {
            float v = acc[i] + bias;
            v = v > 0.f ? v : 0.f;
            partial[i] += v * ow;
        }
    }

#pragma unroll
    for (int i = 0; i < 8; ++i)
        atomicAdd(&red[i + half * 8], partial[i]);   // ds_add_f32
    __syncthreads();

    if (threadIdx.x < 16) {
        int n = r0 + threadIdx.x;
        float v = red[threadIdx.x] + out_b[0];
        int idx;
        if (n < Sc * Bc) { int s = n / Bc, b = n % Bc; idx = b * (Sc + 1) + 1 + s; }
        else             { int b = n - Sc * Bc;        idx = b * (Sc + 1); }
        logits[idx] = v;
    }
}

// ---------------- log-softmax over S+1 = 129 ----------------
__global__ __launch_bounds__(256)
void k_logsoftmax(const float* __restrict__ logits, float* __restrict__ out) {
    __shared__ float sdata[256];
    const int L = Sc + 1;
    int b = blockIdx.x, t = threadIdx.x;
    float v = (t < L) ? logits[(size_t)b * L + t] : -INFINITY;
    sdata[t] = v; __syncthreads();
    for (int s2 = 128; s2 > 0; s2 >>= 1) {
        if (t < s2) sdata[t] = fmaxf(sdata[t], sdata[t + s2]);
        __syncthreads();
    }
    float mx = sdata[0]; __syncthreads();
    float e = (t < L) ? expf(v - mx) : 0.f;
    sdata[t] = e; __syncthreads();
    for (int s2 = 128; s2 > 0; s2 >>= 1) {
        if (t < s2) sdata[t] += sdata[t + s2];
        __syncthreads();
    }
    float lse = logf(sdata[0]);
    if (t < L) out[(size_t)b * L + t] = (v - mx) - lse;
}

// ---------------- host launcher ----------------
extern "C" void kernel_launch(void* const* d_in, const int* in_sizes, int n_in,
                              void* d_out, int out_size, void* d_ws, size_t ws_size,
                              hipStream_t stream) {
    (void)in_sizes; (void)n_in; (void)out_size; (void)ws_size;

    const float* w_ih  = (const float*)d_in[4];
    const float* w_hh  = (const float*)d_in[5];
    const float* b_ih  = (const float*)d_in[6];
    const float* b_hh  = (const float*)d_in[7];
    const float* in_W  = (const float*)d_in[8];
    const float* in_b  = (const float*)d_in[9];
    const float* l0_W  = (const float*)d_in[10];
    const float* l0_b  = (const float*)d_in[11];
    const float* out_W = (const float*)d_in[12];
    const float* out_b = (const float*)d_in[13];

    char* ws = (char*)d_ws;
    size_t off = 0;
    auto take = [&](size_t bytes) -> char* {
        char* p = ws + off;
        off = (off + bytes + 255) & ~(size_t)255;
        return p;
    };
    float*  h_pp   = (float*)take((size_t)2 * Bc * Hc * 4);       // ping-pong hidden fp32
    __bf16* h_ppbf = (__bf16*)take((size_t)2 * Bc * Hc * 2);      // ping-pong hidden bf16
    __bf16* hcat   = (__bf16*)take((size_t)NROWS * Hc * 2);       // relu(h) rows for MLP
    __bf16* xbuf   = (__bf16*)take((size_t)NROWS * Hc * 2);       // relu(embed); reused as a1
    __bf16* xeos   = (__bf16*)take((size_t)Hc * 2);
    __bf16* wihb   = (__bf16*)take((size_t)H3c * Hc * 2);
    __bf16* whhb   = (__bf16*)take((size_t)H3c * Hc * 2);
    __bf16* winb   = (__bf16*)take((size_t)Hc * Hc * 2);
    __bf16* wl0b   = (__bf16*)take((size_t)Hc * Hc * 2);
    float*  logits = (float*)take((size_t)Bc * (Sc + 1) * 4);

    // weight downconverts (L2-resident afterwards)
    k_cvt_bf16<<<(H3c * Hc + 255) / 256, 256, 0, stream>>>(w_ih, wihb, H3c * Hc);
    k_cvt_bf16<<<(H3c * Hc + 255) / 256, 256, 0, stream>>>(w_hh, whhb, H3c * Hc);
    k_cvt_bf16<<<(Hc * Hc + 255) / 256, 256, 0, stream>>>(in_W, winb, Hc * Hc);
    k_cvt_bf16<<<(Hc * Hc + 255) / 256, 256, 0, stream>>>(l0_W, wl0b, Hc * Hc);

    // embeddings
    k_embed_x<<<2048, 256, 0, stream>>>((const int*)d_in[0], (const float*)d_in[3], xbuf);
    k_init_h0_eos<<<(Bc * Hc + 255) / 256, 256, 0, stream>>>(
        (const int*)d_in[1], (const float*)d_in[2], (const float*)d_in[3],
        h_pp, h_ppbf, xeos);

    // 129 sequential GRU steps (last one uses the EOS embedding, broadcast)
    dim3 gstep(Bc / 16, (Hc / 16) / 8);   // (32, 4) blocks of 8 waves
    for (int s = 0; s <= Sc; ++s) {
        const float*  hin    = h_pp   + (size_t)(s & 1) * Bc * Hc;
        const __bf16* hinbf  = h_ppbf + (size_t)(s & 1) * Bc * Hc;
        float*        hout   = h_pp   + (size_t)((s + 1) & 1) * Bc * Hc;
        __bf16*       houtbf = h_ppbf + (size_t)((s + 1) & 1) * Bc * Hc;
        const __bf16* xb;
        int xstride, orow;
        if (s < Sc) { xb = xbuf + (size_t)s * Bc * Hc; xstride = Hc; orow = s * Bc; }
        else        { xb = xeos;                       xstride = 0;  orow = Sc * Bc; }
        k_gru_step<<<gstep, 256, 0, stream>>>(hin, hinbf, hout, houtbf, xb, xstride,
                                              wihb, whhb, b_ih, b_hh, hcat, orow);
    }

    // likelihood MLP over all 66048 rows
    dim3 gin(NROWS / 16, (Hc / 16) / 8);
    k_mlp_in<<<gin, 256, 0, stream>>>(hcat, winb, in_b, xbuf);           // a1 -> xbuf
    k_mlp_out<<<NROWS / 16, 256, 0, stream>>>(xbuf, wl0b, l0_b, out_W, out_b, logits);

    // log-softmax -> d_out [B, S+1]
    k_logsoftmax<<<Bc, 256, 0, stream>>>(logits, (float*)d_out);
}
